// MultiHeadAtt_34325378629732
// MI455X (gfx1250) — compile-verified, matrix operand order checked by reading, not run on hardware
//
#include <hip/hip_runtime.h>
#include <hip/hip_bf16.h>
#include <math.h>

// Problem constants (from the reference)
#define BB 64
#define TT 4096
#define DQ 512
#define DF 256
#define HH 8
#define DD 64
#define EPSV 1e-7f

typedef __attribute__((ext_vector_type(16))) __bf16 v16bf;
typedef __attribute__((ext_vector_type(8)))  float  v8f;

static __device__ __forceinline__ __bf16 f2bf(float f) { return (__bf16)f; }

__global__ __launch_bounds__(256, 2)
void mha_single_q_kernel(const float* __restrict__ query,  // [B, 1, DQ]
                         const float* __restrict__ fact,   // [B, T, DF]
                         const float* __restrict__ Wq,     // [H, DQ, DD]
                         const float* __restrict__ bq,     // [H, DD]
                         const float* __restrict__ Wf,     // [H, DF, DD]
                         const float* __restrict__ bf,     // [H, DD]
                         float* __restrict__ out)          // [B, H*DD]
{
    const int b = blockIdx.x / HH;
    const int h = blockIdx.x % HH;
    const int tid  = threadIdx.x;
    const int wave = tid >> 5;
    const int lane = tid & 31;
    const int half = lane >> 4;   // 0 or 1
    const int lcol = lane & 15;   // column within 16-wide N tile

    // ---------------- LDS ----------------
    __shared__ v16bf wfTiles[32][32];          // 8 K-chunks x 4 N-tiles, B-fragment layout, 32 KB
    __shared__ float fq_lds[DD];
    __shared__ float red_m[8], red_s[8];
    __shared__ float red_o[8][DD];
    __shared__ float red_c[8][DD];

    // ---------------- fq = query @ Wq[h] + bq[h]  (small; ds_add_f32 reduce) ----
    if (tid < DD) fq_lds[tid] = bq[h * DD + tid];
    __syncthreads();
    {
        const int d    = tid & 63;
        const int part = tid >> 6;            // 0..3, each covers 128 of 512 f's
        const float* q  = query + (size_t)b * DQ;
        const float* wq = Wq + (size_t)h * DQ * DD + d;
        float p = 0.f;
        #pragma unroll 4
        for (int f = part * 128; f < part * 128 + 128; ++f)
            p += q[f] * wq[(size_t)f * DD];
        atomicAdd(&fq_lds[d], p);
    }

    // ---------------- Stage Wf[h] -> LDS as bf16 B-fragments ----------------
    // tile(kc,ct): lane slot holds col = ct*16 + slot%16,
    //   K runs: kstart..kstart+7 and kstart+16..kstart+23, kstart = kc*32 + (slot/16)*8
    for (int idx = tid; idx < 32 * 32; idx += 256) {
        const int tile = idx >> 5, slot = idx & 31;
        const int kc = tile >> 2, ct = tile & 3;
        const int col = ct * 16 + (slot & 15);
        const int kstart = kc * 32 + (slot >> 4) * 8;
        const float* wf = Wf + (size_t)h * DF * DD + col;
        v16bf v;
        #pragma unroll
        for (int e = 0; e < 8; ++e) {
            v[e]     = f2bf(wf[(size_t)(kstart + e) * DD]);
            v[e + 8] = f2bf(wf[(size_t)(kstart + 16 + e) * DD]);
        }
        wfTiles[tile][slot] = v;
    }
    __syncthreads();

    // Per-lane broadcast constants
    float fqv[4], bfv[4];
    #pragma unroll
    for (int ct = 0; ct < 4; ++ct) {
        fqv[ct] = fq_lds[ct * 16 + lcol];
        bfv[ct] = bf[h * DD + ct * 16 + lcol];
    }

    // ---------------- Flash loop over T (16-row tiles, 8 waves round-robin) ----
    float m_run = -3.0e38f, s_run = 0.f;
    float outp[4] = {0.f, 0.f, 0.f, 0.f};
    float csum[4] = {0.f, 0.f, 0.f, 0.f};

    for (int tile = wave; tile < TT / 16; tile += 8) {
        const int row = tile * 16 + lcol;                       // A-matrix row = lane%16
        const float* arow = fact + ((size_t)b * TT + row) * DF; // 1 KB aligned

        if (tile + 8 < TT / 16) {
            const float* nxt = fact + ((size_t)b * TT + (tile + 8) * 16 + lcol) * DF;
            __builtin_prefetch(nxt, 0, 1);                      // global_prefetch_b8
        }

        v8f acc[4];
        #pragma unroll
        for (int ct = 0; ct < 4; ++ct)
            #pragma unroll
            for (int j = 0; j < 8; ++j) acc[ct][j] = 0.f;

        #pragma unroll
        for (int kc = 0; kc < 8; ++kc) {
            const int kb = kc * 32 + half * 8;                  // ISA 16-bit A layout
            const float4 a0 = *(const float4*)(arow + kb);
            const float4 a1 = *(const float4*)(arow + kb + 4);
            const float4 a2 = *(const float4*)(arow + kb + 16);
            const float4 a3 = *(const float4*)(arow + kb + 20);
            v16bf af;
            af[0]  = f2bf(a0.x); af[1]  = f2bf(a0.y); af[2]  = f2bf(a0.z); af[3]  = f2bf(a0.w);
            af[4]  = f2bf(a1.x); af[5]  = f2bf(a1.y); af[6]  = f2bf(a1.z); af[7]  = f2bf(a1.w);
            af[8]  = f2bf(a2.x); af[9]  = f2bf(a2.y); af[10] = f2bf(a2.z); af[11] = f2bf(a2.w);
            af[12] = f2bf(a3.x); af[13] = f2bf(a3.y); af[14] = f2bf(a3.z); af[15] = f2bf(a3.w);

            #pragma unroll
            for (int ct = 0; ct < 4; ++ct) {
                acc[ct] = __builtin_amdgcn_wmma_f32_16x16x32_bf16(
                    false, af, false, wfTiles[kc * 4 + ct][lane],
                    (short)0, acc[ct], false, false);
            }
        }

        // ff = raw + bf  (bias participates in scores, output, and colsum)
        #pragma unroll
        for (int ct = 0; ct < 4; ++ct)
            #pragma unroll
            for (int j = 0; j < 8; ++j) acc[ct][j] += bfv[ct];

        // scores: row m = j + 8*half; reduce partial dot across the 16 lanes of the half
        float p[8];
        #pragma unroll
        for (int j = 0; j < 8; ++j)
            p[j] = acc[0][j] * fqv[0] + acc[1][j] * fqv[1]
                 + acc[2][j] * fqv[2] + acc[3][j] * fqv[3];
        #pragma unroll
        for (int msk = 1; msk < 16; msk <<= 1)
            #pragma unroll
            for (int j = 0; j < 8; ++j) p[j] += __shfl_xor(p[j], msk, 32);

        // online softmax (wave-uniform m and s; halves merged via xor-16)
        float tm = p[0];
        #pragma unroll
        for (int j = 1; j < 8; ++j) tm = fmaxf(tm, p[j]);
        tm = fmaxf(tm, __shfl_xor(tm, 16, 32));
        const float m_new = fmaxf(m_run, tm);
        const float scale = __expf(m_run - m_new);
        float e[8], ps = 0.f;
        #pragma unroll
        for (int j = 0; j < 8; ++j) { e[j] = __expf(p[j] - m_new); ps += e[j]; }
        ps += __shfl_xor(ps, 16, 32);
        s_run = s_run * scale + ps;
        m_run = m_new;

        #pragma unroll
        for (int ct = 0; ct < 4; ++ct) {
            float ae = 0.f, ac = 0.f;
            #pragma unroll
            for (int j = 0; j < 8; ++j) { ae += e[j] * acc[ct][j]; ac += acc[ct][j]; }
            outp[ct] = outp[ct] * scale + ae;   // this half's rows only
            csum[ct] += ac;
        }
    }

    // merge the two row-halves (same d columns live in lanes l and l+16)
    #pragma unroll
    for (int ct = 0; ct < 4; ++ct) {
        outp[ct] += __shfl_xor(outp[ct], 16, 32);
        csum[ct] += __shfl_xor(csum[ct], 16, 32);
    }

    // ---------------- cross-wave combine ----------------
    if (lane == 0) { red_m[wave] = m_run; red_s[wave] = s_run; }
    if (lane < 16) {
        #pragma unroll
        for (int ct = 0; ct < 4; ++ct) {
            red_o[wave][ct * 16 + lane] = outp[ct];
            red_c[wave][ct * 16 + lane] = csum[ct];
        }
    }
    __syncthreads();

    if (tid < DD) {
        float M = red_m[0];
        #pragma unroll
        for (int w = 1; w < 8; ++w) M = fmaxf(M, red_m[w]);
        float den = 0.f, num = 0.f, cs = 0.f;
        #pragma unroll
        for (int w = 0; w < 8; ++w) {
            const float sc = __expf(red_m[w] - M);
            den += sc * red_s[w];
            num += sc * red_o[w][tid];
            cs  += red_c[w][tid];
        }
        out[(size_t)b * (HH * DD) + h * DD + tid] = num / den + EPSV * cs;
    }
}

extern "C" void kernel_launch(void* const* d_in, const int* in_sizes, int n_in,
                              void* d_out, int out_size, void* d_ws, size_t ws_size,
                              hipStream_t stream) {
    const float* query = (const float*)d_in[0];
    const float* fact  = (const float*)d_in[1];
    const float* Wq    = (const float*)d_in[2];
    const float* bq    = (const float*)d_in[3];
    const float* Wf    = (const float*)d_in[4];
    const float* bf    = (const float*)d_in[5];
    float* out = (float*)d_out;
    (void)in_sizes; (void)n_in; (void)out_size; (void)d_ws; (void)ws_size;

    dim3 grid(BB * HH);
    dim3 block(256);
    hipLaunchKernelGGL(mha_single_q_kernel, grid, block, 0, stream,
                       query, fact, Wq, bq, Wf, bf, out);
}
